// ODEBlock_4140348473706
// MI455X (gfx1250) — compile-verified, matrix operand order checked by reading, not run on hardware
//
#include <hip/hip_runtime.h>
#include <math.h>

// ---------------------------------------------------------------------------
// Neural ODE block: y' = tanh(y W1 + b1) W2 + b2,  t in [0, 5]
// Fixed-step RK4 (40 steps) replacing adaptive dopri5 (graph-capture safe).
// GEMMs on the CDNA5 WMMA pipe (v_wmma_f32_16x16x32_f16, fp32 accum) with
// double-buffered global_load_async_to_lds_b128 staging (ASYNCcnt pipeline).
// ---------------------------------------------------------------------------

typedef __attribute__((ext_vector_type(16))) _Float16 v16h;
typedef __attribute__((ext_vector_type(8)))  float    v8f;

#define BATCH  8192
#define DIM    512
#define NSTEPS 40

#define BM 128
#define BN 128
#define BK 32      // == WMMA K for f16
#define LDT 40     // LDS row stride in halves: 80 B rows (16B-aligned chunks)
#define KTILES (DIM / BK)

union HVec { v16h h; uint4 u[2]; };

// Fragment load from LDS tile (row-major, stride LDT). gfx1250 16-bit A/B
// layout: lanes 0-15 hold K kb+0..7 and kb+16..23 (kb=0); lanes 16-31 kb=8.
__device__ __forceinline__ v16h ld_frag(const _Float16* rowBase, int kb) {
    HVec f;
    f.u[0] = *(const uint4*)(rowBase + kb);
    f.u[1] = *(const uint4*)(rowBase + kb + 16);
    return f.h;
}

// Issue this thread's 2 async 16B copies of one (rows x BK) tile slice:
// memory -> LDS directly, tracked by ASYNCcnt (no VGPR round-trip).
__device__ __forceinline__ void issue_async_tile(const _Float16* __restrict__ G,
                                                 _Float16* lds, int base_row,
                                                 int k0, int tid) {
#pragma unroll
    for (int t = 0; t < 2; ++t) {
        const int c   = tid + t * 256;   // 512 chunks of 8 halves
        const int row = c >> 2;          // 4 chunks per row
        const int col = (c & 3) * 8;
        const _Float16* gp = &G[(size_t)(base_row + row) * DIM + k0 + col];
        const unsigned  lp = (unsigned)(size_t)&lds[row * LDT + col];
        asm volatile("global_load_async_to_lds_b128 %0, %1, off"
                     :: "v"(lp), "v"(gp) : "memory");
    }
}

// Shared GEMM core: C[BMxBN] = A[BMxDIM] * Bt[BNxDIM]^T  (Bt is [N][K] f16).
// 8 waves: 4 (M) x 2 (N); each wave owns a 32x64 block = 2x4 WMMA tiles.
__device__ __forceinline__ void gemm_core(const _Float16* __restrict__ A,
                                          const _Float16* __restrict__ Bt,
                                          v8f acc[2][4]) {
    __shared__ _Float16 As[2][BM * LDT];
    __shared__ _Float16 Bs[2][BN * LDT];

    const int tid  = threadIdx.x;
    const int lane = tid & 31;
    const int wave = tid >> 5;      // 0..7
    const int wm   = wave >> 1;     // 0..3 -> 32 rows
    const int wn   = wave & 1;      // 0..1 -> 64 cols

    const int m0 = blockIdx.x * BM;
    const int n0 = blockIdx.y * BN;

    v8f z = {};
#pragma unroll
    for (int i = 0; i < 2; ++i)
#pragma unroll
        for (int j = 0; j < 4; ++j)
            acc[i][j] = z;

    const int r  = lane & 15;
    const int kb = (lane >> 4) * 8;

    // Prologue: stage tile 0 into buffer 0.
    issue_async_tile(A,  As[0], m0, 0, tid);
    issue_async_tile(Bt, Bs[0], n0, 0, tid);

    for (int kt = 0; kt < KTILES; ++kt) {
        const int cur = kt & 1;
        if (kt + 1 < KTILES) {
            // Stage next tile into the other buffer while we compute.
            issue_async_tile(A,  As[cur ^ 1], m0, (kt + 1) * BK, tid);
            issue_async_tile(Bt, Bs[cur ^ 1], n0, (kt + 1) * BK, tid);
            // Async loads complete in order: <=4 outstanding => tile kt landed.
            asm volatile("s_wait_asynccnt 0x4" ::: "memory");
        } else {
            asm volatile("s_wait_asynccnt 0x0" ::: "memory");
        }
        __syncthreads();

        v16h afrag[2], bfrag[4];
#pragma unroll
        for (int i = 0; i < 2; ++i)
            afrag[i] = ld_frag(&As[cur][(wm * 32 + i * 16 + r) * LDT], kb);
#pragma unroll
        for (int j = 0; j < 4; ++j)
            bfrag[j] = ld_frag(&Bs[cur][(wn * 64 + j * 16 + r) * LDT], kb);

#pragma unroll
        for (int i = 0; i < 2; ++i)
#pragma unroll
            for (int j = 0; j < 4; ++j)
                acc[i][j] = __builtin_amdgcn_wmma_f32_16x16x32_f16(
                    false, afrag[i], false, bfrag[j],
                    (short)0, acc[i][j], false, false);

        __syncthreads();  // protect buffer cur before it is re-staged at kt+2
    }
}

// GEMM 1: H = tanh(A @ W1 + b1), f16 output.
__global__ __launch_bounds__(256)
void gemm_tanh(const _Float16* __restrict__ A, const _Float16* __restrict__ W1t,
               const float* __restrict__ b1, _Float16* __restrict__ H) {
    v8f acc[2][4];
    gemm_core(A, W1t, acc);

    const int lane  = threadIdx.x & 31;
    const int wave  = threadIdx.x >> 5;
    const int wm    = wave >> 1, wn = wave & 1;
    const int m0    = blockIdx.x * BM, n0 = blockIdx.y * BN;
    const int rbase = (lane >> 4) * 8;
    const int c16   = lane & 15;

#pragma unroll
    for (int i = 0; i < 2; ++i)
#pragma unroll
        for (int j = 0; j < 4; ++j) {
            const int colg = n0 + wn * 64 + j * 16 + c16;
            const float bv = b1[colg];
#pragma unroll
            for (int e = 0; e < 8; ++e) {
                const int rowg = m0 + wm * 32 + i * 16 + rbase + e;
                H[(size_t)rowg * DIM + colg] = (_Float16)tanhf(acc[i][j][e] + bv);
            }
        }
}

// GEMM 2: k = H @ W2 + b2, fused RK4 epilogue:
//   accout = accin + wgt * k
//   if writeA: Anext = f16(y + cnext * k)
__global__ __launch_bounds__(256)
void gemm_stage(const _Float16* __restrict__ Hb, const _Float16* __restrict__ W2t,
                const float* __restrict__ b2,
                const float* __restrict__ y, const float* __restrict__ accin,
                float* __restrict__ accout, _Float16* __restrict__ Anext,
                float wgt, float cnext, int writeA) {
    v8f acc[2][4];
    gemm_core(Hb, W2t, acc);

    const int lane  = threadIdx.x & 31;
    const int wave  = threadIdx.x >> 5;
    const int wm    = wave >> 1, wn = wave & 1;
    const int m0    = blockIdx.x * BM, n0 = blockIdx.y * BN;
    const int rbase = (lane >> 4) * 8;
    const int c16   = lane & 15;

#pragma unroll
    for (int i = 0; i < 2; ++i)
#pragma unroll
        for (int j = 0; j < 4; ++j) {
            const int colg = n0 + wn * 64 + j * 16 + c16;
            const float bv = b2[colg];
#pragma unroll
            for (int e = 0; e < 8; ++e) {
                const int rowg = m0 + wm * 32 + i * 16 + rbase + e;
                const size_t idx = (size_t)rowg * DIM + colg;
                const float kval = acc[i][j][e] + bv;
                accout[idx] = accin[idx] + wgt * kval;
                if (writeA)
                    Anext[idx] = (_Float16)(y[idx] + cnext * kval);
            }
        }
}

// W (row-major [K][N] f32) -> Wt ([N][K] f16) so B tiles are contiguous in K.
__global__ __launch_bounds__(256)
void transpose_f16(const float* __restrict__ W, _Float16* __restrict__ Wt) {
    int idx = blockIdx.x * 256 + threadIdx.x;
    int k = idx / DIM, n = idx % DIM;
    Wt[(size_t)n * DIM + k] = (_Float16)W[(size_t)k * DIM + n];
}

__global__ __launch_bounds__(256)
void init_state(const float* __restrict__ x, float* __restrict__ y,
                _Float16* __restrict__ A) {
    int idx = blockIdx.x * 256 + threadIdx.x;
    float v = x[idx];
    y[idx] = v;
    A[idx] = (_Float16)v;
}

__global__ __launch_bounds__(256)
void copy_out(const float* __restrict__ y, float* __restrict__ out) {
    int idx = blockIdx.x * 256 + threadIdx.x;
    out[idx] = y[idx];
}

extern "C" void kernel_launch(void* const* d_in, const int* in_sizes, int n_in,
                              void* d_out, int out_size, void* d_ws, size_t ws_size,
                              hipStream_t stream) {
    const float* x  = (const float*)d_in[0];
    const float* W1 = (const float*)d_in[1];
    const float* b1 = (const float*)d_in[2];
    const float* W2 = (const float*)d_in[3];
    const float* b2 = (const float*)d_in[4];
    float* out = (float*)d_out;

    // Workspace carve-up
    char* ws = (char*)d_ws;
    size_t off = 0;
    auto alloc = [&](size_t bytes) -> void* {
        void* p = ws + off;
        off += (bytes + 255) & ~(size_t)255;
        return p;
    };
    _Float16* W1t = (_Float16*)alloc((size_t)DIM * DIM * sizeof(_Float16));
    _Float16* W2t = (_Float16*)alloc((size_t)DIM * DIM * sizeof(_Float16));
    _Float16* A   = (_Float16*)alloc((size_t)BATCH * DIM * sizeof(_Float16));
    _Float16* Hb  = (_Float16*)alloc((size_t)BATCH * DIM * sizeof(_Float16));
    float*    y   = (float*)alloc((size_t)BATCH * DIM * sizeof(float));
    float*    acc = (float*)alloc((size_t)BATCH * DIM * sizeof(float));

    const dim3 blk(256);
    transpose_f16<<<(DIM * DIM) / 256, blk, 0, stream>>>(W1, W1t);
    transpose_f16<<<(DIM * DIM) / 256, blk, 0, stream>>>(W2, W2t);
    init_state<<<(BATCH * DIM) / 256, blk, 0, stream>>>(x, y, A);

    const dim3 gg(BATCH / BM, DIM / BN);  // 64 x 4 workgroups
    const float h = 5.0f / (float)NSTEPS;

    for (int s = 0; s < NSTEPS; ++s) {
        // k1
        gemm_tanh <<<gg, blk, 0, stream>>>(A, W1t, b1, Hb);
        gemm_stage<<<gg, blk, 0, stream>>>(Hb, W2t, b2, y, y,   acc, A,      h / 6.0f, h * 0.5f, 1);
        // k2
        gemm_tanh <<<gg, blk, 0, stream>>>(A, W1t, b1, Hb);
        gemm_stage<<<gg, blk, 0, stream>>>(Hb, W2t, b2, y, acc, acc, A,      h / 3.0f, h * 0.5f, 1);
        // k3
        gemm_tanh <<<gg, blk, 0, stream>>>(A, W1t, b1, Hb);
        gemm_stage<<<gg, blk, 0, stream>>>(Hb, W2t, b2, y, acc, acc, A,      h / 3.0f, h,        1);
        // k4 -> y_{n+1}
        gemm_tanh <<<gg, blk, 0, stream>>>(A, W1t, b1, Hb);
        gemm_stage<<<gg, blk, 0, stream>>>(Hb, W2t, b2, y, acc, y,   nullptr, h / 6.0f, 0.0f,    0);
    }

    copy_out<<<(BATCH * DIM) / 256, blk, 0, stream>>>(y, out);
}
